// S_VAE_COUPLE_53480932770640
// MI455X (gfx1250) — matrix operand order repaired and checked by reading, not compiled, not run on hardware
//
#include <hip/hip_runtime.h>
#include <stdint.h>

typedef __bf16 bf16;
typedef __attribute__((ext_vector_type(16))) bf16 v16bf;
typedef __attribute__((ext_vector_type(8)))  float v8f;

#define IN_DIM 1024
#define MDIM   64
#define WT_STRIDE 1040      // 1024 + 16 pad: keeps 32B alignment, staggers LDS banks
#define TILES_PER_BLOCK 2   // row-tiles (of 128 rows) per block; amortizes W staging

// ============================================================================
// Kernel 1: fused GEMM (WMMA bf16) + row-normalize (miu) + kappa GEMV+softplus
//   block = 256 threads = 8 waves; each wave: 16 rows x 64 cols per tile
// ============================================================================
__global__ __launch_bounds__(256) void vmf_head_kernel(
    const float* __restrict__ x, const float* __restrict__ W_miu,
    const float* __restrict__ b_miu, const float* __restrict__ W_kappa,
    const float* __restrict__ b_kappa, float* __restrict__ miu_out,
    float* __restrict__ kappa_out) {
  extern __shared__ __align__(32) char smem[];
  bf16*  Wt = (bf16*)smem;                                        // [64][WT_STRIDE] col-major bf16
  float* Wk = (float*)(smem + (size_t)MDIM * WT_STRIDE * sizeof(bf16));

  const int tid = threadIdx.x;
  // Stage W_miu transposed into LDS (bf16) + W_kappa (fp32), once per block
  for (int i = tid; i < IN_DIM * MDIM; i += 256) {
    int k = i >> 6, n = i & 63;
    Wt[n * WT_STRIDE + k] = (bf16)W_miu[i];
  }
  for (int i = tid; i < IN_DIM; i += 256) Wk[i] = W_kappa[i];
  __syncthreads();

  const int lane = tid & 31;
  const int wave = tid >> 5;
  const int l15  = lane & 15;
  const int half = lane >> 4;                 // 0: low 16 lanes, 1: high 16 lanes
  const float bk0 = b_kappa[0];
  float bm[4];
#pragma unroll
  for (int t = 0; t < 4; ++t) bm[t] = b_miu[t * 16 + l15];

  // LDS byte base for this lane's B-fragment columns
  const bf16* wtLane[4];
#pragma unroll
  for (int t = 0; t < 4; ++t)
    wtLane[t] = Wt + (size_t)(t * 16 + l15) * WT_STRIDE + half * 16;

  for (int tile = 0; tile < TILES_PER_BLOCK; ++tile) {
    const int rowBase = (blockIdx.x * TILES_PER_BLOCK + tile) * 128 + wave * 16;
    const float* xrow = x + (size_t)(rowBase + l15) * IN_DIM;

    v8f acc[4] = {};                          // 4 N-tiles of 16x16 f32
    float kacc0 = 0.0f, kacc1 = 0.0f;         // two chains for the kappa GEMV

#pragma unroll 2
    for (int k0 = 0; k0 < IN_DIM; k0 += 32) {
      // ---- A fragment (16x32 bf16), contiguous K order (B uses the same):
      //   lanes 0-15  : row=l15, K = {k0..k0+15}
      //   lanes 16-31 : row=l15, K = {k0+16..k0+31}
      const int kb = k0 + half * 16;
      float xf[16];
      *(float4*)(xf + 0)  = *(const float4*)(xrow + kb);
      *(float4*)(xf + 4)  = *(const float4*)(xrow + kb + 4);
      *(float4*)(xf + 8)  = *(const float4*)(xrow + kb + 8);
      *(float4*)(xf + 12) = *(const float4*)(xrow + kb + 12);
      __builtin_prefetch(xrow + k0 + 64, 0, 1);  // global_prefetch_b8 next tile

      // ---- issue all 4 B-fragment loads up front so ds_loads pipeline
      v16bf bfrag[4];
#pragma unroll
      for (int t = 0; t < 4; ++t)
        bfrag[t] = *(const v16bf*)(wtLane[t] + k0);

      v16bf afrag;
#pragma unroll
      for (int j = 0; j < 16; ++j) afrag[j] = (bf16)xf[j];

      // fused kappa GEMV on the fp32 values already in registers (2 chains =
      // independent VALU work to co-issue with the WMMA/XDL pipe)
#pragma unroll
      for (int j = 0; j < 8; ++j) {
        kacc0 = fmaf(xf[j],     Wk[kb + j],      kacc0);
        kacc1 = fmaf(xf[8 + j], Wk[kb + 8 + j], kacc1);
      }

#pragma unroll
      for (int t = 0; t < 4; ++t)
        acc[t] = __builtin_amdgcn_wmma_f32_16x16x32_bf16(
            false, afrag, false, bfrag[t], (short)0, acc[t], false, false);
    }

    // ---- kappa: lanes L and L^16 hold disjoint K-halves of the same row
    float kacc = kacc0 + kacc1;
    kacc += __shfl_xor(kacc, 16, 32);
    {
      float klin = kacc + bk0;
      float sp = (klin > 20.0f) ? klin : log1pf(__expf(klin));
      if (lane < 16) kappa_out[rowBase + lane] = sp + 1.0f;
    }

    // ---- bias + row L2 normalize + store
#pragma unroll
    for (int r = 0; r < 8; ++r) {
      float s = 0.0f;
#pragma unroll
      for (int t = 0; t < 4; ++t) {
        acc[t][r] += bm[t];
        s += acc[t][r] * acc[t][r];
      }
#pragma unroll
      for (int o = 8; o > 0; o >>= 1) s += __shfl_xor(s, o, 32);  // per-16-lane half
      float rn = rsqrtf(fmaxf(s, 1e-30f));
      const int row = rowBase + r + half * 8;   // C layout: lanes>=16 hold M=r+8
#pragma unroll
      for (int t = 0; t < 4; ++t)
        miu_out[(size_t)row * MDIM + t * 16 + l15] = acc[t][r] * rn;
    }
  }
}

// ============================================================================
// Kernel 2: vMF rejection sampling + tangent draw + Householder (1 wave / row)
// ============================================================================
__device__ __forceinline__ unsigned hash3(unsigned a, unsigned b, unsigned c) {
  unsigned x = a * 0x9E3779B1u + b * 0x85EBCA77u + c * 0xC2B2AE3Du + 0x27D4EB2Fu;
  x ^= x >> 15; x *= 0x2C1B3C6Du;
  x ^= x >> 12; x *= 0x297A2D39u;
  x ^= x >> 15;
  return x;
}
__device__ __forceinline__ float u01(unsigned h) {
  return (float)(h >> 8) * (1.0f / 16777216.0f) + (0.5f / 16777216.0f);
}
__device__ __forceinline__ float wsum(float v) {
#pragma unroll
  for (int o = 16; o > 0; o >>= 1) v += __shfl_xor(v, o, 32);
  return v;
}
// Gamma(31.5) via Marsaglia-Tsang (accept rate ~99%), counter-based RNG
__device__ float gamma_a(unsigned row, unsigned sid) {
  const float d  = 31.5f - (1.0f / 3.0f);
  const float cc = 1.0f / (3.0f * sqrtf(d));
  for (unsigned it = 0; it < 8; ++it) {
    unsigned base = sid * 32u + it * 3u;
    float u1 = u01(hash3(row, base + 0u, 0x68BC21EBu));
    float u2 = u01(hash3(row, base + 1u, 0x68BC21EBu));
    float r  = sqrtf(fmaxf(-2.0f * __logf(u1), 0.0f));
    float xn = r * __cosf(6.28318530718f * u2);
    float v  = 1.0f + cc * xn;
    if (v <= 0.0f) continue;
    v = v * v * v;
    float u3 = u01(hash3(row, base + 2u, 0x9D2C5681u));
    if (__logf(u3) < 0.5f * xn * xn + d - d * v + d * __logf(v)) return d * v;
  }
  return d;
}
__device__ __forceinline__ bool propose(unsigned row, unsigned p, float bb,
                                        float a, float dpar, float& w_out) {
  float g1 = gamma_a(row, p * 4u + 0u);
  float g2 = gamma_a(row, p * 4u + 1u);
  float e  = g1 / (g1 + g2);                       // Beta(31.5, 31.5)
  float u  = fmaxf(u01(hash3(row, p * 4u + 2u, 0x85EBCA6Bu)), 1e-7f);
  float denom = 1.0f - (1.0f - bb) * e;
  w_out = (1.0f - (1.0f + bb) * e) / denom;
  float t = 2.0f * a * bb / denom;
  return (63.0f * __logf(t) - t + dpar) > __logf(u);
}

__global__ __launch_bounds__(256) void vmf_sample_kernel(
    const float* __restrict__ kappa_buf, float* __restrict__ zio) {
  const int lane = threadIdx.x & 31;
  const unsigned row = blockIdx.x * 8u + (threadIdx.x >> 5);
  const float kap = kappa_buf[row];

  // Ulrich/Davidson envelope, m = 64
  float c      = sqrtf(4.0f * kap * kap + 63.0f * 63.0f);
  float b_true = (c - 2.0f * kap) / 63.0f;
  float b_app  = 63.0f / (4.0f * kap);
  float sg     = fminf(fmaxf(kap - 10.0f, 0.0f), 1.0f);
  float bb     = b_app * sg + b_true * (1.0f - sg);
  float a      = (63.0f + 2.0f * kap + c) * 0.25f;
  float dpar   = 4.0f * a * bb / (1.0f + bb) - 63.0f * __logf(63.0f);

  // 32 proposals in parallel per round; first-accept via ballot+ffs
  float w0cand, w;
  bool acc0 = propose(row, (unsigned)lane, bb, a, dpar, w0cand);
  unsigned long long m0 = __ballot(acc0);
  if (m0) {
    w = __shfl(w0cand, __ffsll(m0) - 1, 32);
  } else {
    float w1cand;
    bool acc1 = propose(row, (unsigned)lane + 32u, bb, a, dpar, w1cand);
    unsigned long long m1 = __ballot(acc1);
    w = m1 ? __shfl(w1cand, __ffsll(m1) - 1, 32)
           : __shfl(w0cand, 0, 32);               // argmax of all-false -> idx 0
  }

  // tangent direction: 63 normals, 2 components per lane (dims lane, lane+32)
  const int n0 = lane, n1 = lane + 32;
  float uu1 = u01(hash3(row, 300u + lane, 0x51633E2Du));
  float uu2 = u01(hash3(row, 340u + lane, 0x51633E2Du));
  float rr  = sqrtf(fmaxf(-2.0f * __logf(uu1), 0.0f));
  float v0  = (n0 == 0) ? 0.0f : rr * __cosf(6.28318530718f * uu2);
  float v1  = rr * __sinf(6.28318530718f * uu2);
  float vin = rsqrtf(fmaxf(wsum(v0 * v0 + v1 * v1), 1e-20f));
  float wperp = sqrtf(fmaxf(1.0f - w * w, 1e-10f));
  float x0 = (n0 == 0) ? w : wperp * v0 * vin;
  float x1 = wperp * v1 * vin;

  // Householder e1 -> miu (miu read from d_out, z written in place)
  float mu0 = zio[(size_t)row * MDIM + n0];
  float mu1 = zio[(size_t)row * MDIM + n1];
  float uh0 = ((n0 == 0) ? 1.0f : 0.0f) - mu0;
  float uh1 = -mu1;
  float uhin = 1.0f / (sqrtf(wsum(uh0 * uh0 + uh1 * uh1)) + 1e-5f);
  uh0 *= uhin; uh1 *= uhin;
  float dot = wsum(x0 * uh0 + x1 * uh1);
  zio[(size_t)row * MDIM + n0] = x0 - 2.0f * dot * uh0;
  zio[(size_t)row * MDIM + n1] = x1 - 2.0f * dot * uh1;
}

// ============================================================================
extern "C" void kernel_launch(void* const* d_in, const int* in_sizes, int n_in,
                              void* d_out, int out_size, void* d_ws, size_t ws_size,
                              hipStream_t stream) {
  const float* x       = (const float*)d_in[0];
  const float* W_miu   = (const float*)d_in[1];
  const float* b_miu   = (const float*)d_in[2];
  const float* W_kappa = (const float*)d_in[3];
  const float* b_kappa = (const float*)d_in[4];
  float* out   = (float*)d_out;
  float* kappa = (float*)d_ws;   // B floats = 256 KB scratch

  const int B = in_sizes[0] / IN_DIM;
  const size_t shmem = (size_t)MDIM * WT_STRIDE * sizeof(bf16) + IN_DIM * sizeof(float);

  hipLaunchKernelGGL(vmf_head_kernel, dim3(B / (128 * TILES_PER_BLOCK)), dim3(256),
                     shmem, stream, x, W_miu, b_miu, W_kappa, b_kappa, out, kappa);
  hipLaunchKernelGGL(vmf_sample_kernel, dim3(B / 8), dim3(256), 0, stream,
                     kappa, out);
}
